// MusicGenerationV2_6786048328588
// MI455X (gfx1250) — matrix operand order repaired and checked by reading, not compiled, not run on hardware
//
#include <hip/hip_runtime.h>

typedef _Float16 f16;
typedef f16   v8h  __attribute__((ext_vector_type(8)));
typedef f16   v16h __attribute__((ext_vector_type(16)));
typedef float v8f  __attribute__((ext_vector_type(8)));

#define T_STEPS 128
#define HID     64
#define NB      12288          // 12 notes * 1024 batch
#define IN1     24
#define K2      8192           // T_STEPS*HID
#define SPLIT   4              // waves cooperating on one 16-row tile (t-loop split)
#define WAVES_PER_BLOCK 8
#define GROUPS_PER_BLOCK (WAVES_PER_BLOCK / SPLIT)   // 2 row-tiles per block
#define ROW_TILES (NB / 16)                          // 768
#define NUM_BLOCKS (ROW_TILES / GROUPS_PER_BLOCK)    // 384

// workspace layout (bytes)
#define W2FRAG_ELEMS (256 * 6 * 32 * 16)   // 786432 halves (= 96*8192)
#define W1FRAG_ELEMS (12 * 32 * 16)        // 6144 halves
#define OFF_W1FRAG  (W2FRAG_ELEMS * 2)
#define OFF_B1      (OFF_W1FRAG + W1FRAG_ELEMS * 2)
#define OFF_B2      (OFF_B1 + 256 * 4)

#define LOG2E 1.44269504088896340736f

// v_exp_f32 + v_rcp_f32 based activations (no IEEE divide chains).
// Saturation: x->+inf => sigmoid->1, tanh->1 ; x->-inf => 0 / -1.  Correct.
__device__ __forceinline__ float fast_sigmoid(float x) {
    return __builtin_amdgcn_rcpf(1.0f + __builtin_amdgcn_exp2f(-x * LOG2E));
}
__device__ __forceinline__ float fast_tanh(float x) {
    return 1.0f - 2.0f * __builtin_amdgcn_rcpf(1.0f + __builtin_amdgcn_exp2f(x * (2.0f * LOG2E)));
}

// ---------------------------------------------------------------------------
// Prep: bias sums + W_ih_t re-swizzled into 12 B-fragments (i,g,o gates only;
// f-gate is dead since c_prev == 0). B frag layout: lane holds col N=lane%16,
// halves e -> K = (lane<16?0:16)+e ; K>=24 padded with 0.
// ---------------------------------------------------------------------------
__global__ void prep_small(const float* __restrict__ Wih_t,
                           const float* __restrict__ bih_t, const float* __restrict__ bhh_t,
                           const float* __restrict__ bih_n, const float* __restrict__ bhh_n,
                           f16* __restrict__ w1f, float* __restrict__ b1s, float* __restrict__ b2s)
{
    const int tid = threadIdx.x;
    if (tid < 256) b1s[tid] = bih_t[tid] + bhh_t[tid];
    if (tid < 96)  b2s[tid] = bih_n[tid] + bhh_n[tid];
    for (int idx = tid; idx < W1FRAG_ELEMS; idx += 256) {
        const int e    = idx & 15;
        const int lane = (idx >> 4) & 31;
        const int f    = idx >> 9;          // 0..11  (3 gates x 4 hTiles)
        const int grpI = f >> 2;            // 0=i, 1=g, 2=o
        const int ht   = f & 3;
        const int gbase = (grpI == 0) ? 0 : ((grpI == 1) ? 128 : 192);
        const int ncol = gbase + ht * 16 + (lane & 15);
        const int k    = ((lane < 16) ? 0 : 16) + e;
        const float v  = (k < IN1) ? Wih_t[ncol * IN1 + k] : 0.0f;
        w1f[idx] = (f16)v;
    }
}

// W_ih_n [96,8192] -> fragment-ordered f16: [ks(256)][tile(6)][lane(32)][e(16)]
__global__ void prep_w2(const float* __restrict__ Wih_n, f16* __restrict__ w2f)
{
    long idx = (long)blockIdx.x * blockDim.x + threadIdx.x;
    const long stride = (long)gridDim.x * blockDim.x;
    for (; idx < (long)W2FRAG_ELEMS; idx += stride) {
        const int e    = (int)(idx & 15);
        const int lane = (int)((idx >> 4) & 31);
        const int f    = (int)((idx >> 9) % 6);
        const int ks   = (int)(idx / 3072);
        const int n    = f * 16 + (lane & 15);
        const int k    = ks * 32 + ((lane < 16) ? 0 : 16) + e;
        w2f[idx] = (f16)Wih_n[(long)n * K2 + k];
    }
}

// ---------------------------------------------------------------------------
// Fused kernel: per wave = one 16-row (nb) tile, 1/SPLIT of the t-loop.
// Per t: stage-1 (A from x, 12 WMMA, activations) -> LDS -> two stage-2
// A-fragments -> 12 WMMA into persistent f32 accumulators. Epilogue reduces
// the SPLIT partial accumulators in LDS and applies the note-cell activation
// + 0.5 threshold.
// ---------------------------------------------------------------------------
__global__ __launch_bounds__(256) void music_fused(
    const float* __restrict__ x,
    const f16*   __restrict__ w2f,
    const f16*   __restrict__ w1f,
    const float* __restrict__ b1s,
    const float* __restrict__ b2s,
    float*       __restrict__ out)
{
    __shared__ __align__(16) f16 stage[WAVES_PER_BLOCK][16][64];
    __shared__ float gates[GROUPS_PER_BLOCK][16][96];

    const int tid  = threadIdx.x;
    const int wave = tid >> 5;
    const int lane = tid & 31;
    const int grp  = wave / SPLIT;               // row-tile group in block
    const int s    = wave % SPLIT;               // t-split index
    const int groupGlobal = blockIdx.x * GROUPS_PER_BLOCK + grp;
    const int nb0  = groupGlobal * 16;
    const int halfSel = (lane < 16) ? 0 : 1;
    const int mrow = lane & 15;

    // init group accumulator with (b_ih_n + b_hh_n)
    if (s == 0) {
        for (int idx = lane; idx < 16 * 96; idx += 32)
            gates[grp][idx / 96][idx % 96] = b2s[idx % 96];
    }
    __syncthreads();

    v8f zero = {};
    v8f acc[6];
#pragma unroll
    for (int f = 0; f < 6; ++f) acc[f] = zero;

    const long rowNb = (long)nb0 + mrow;

#pragma unroll 1
    for (int tt = 0; tt < T_STEPS / SPLIT; ++tt) {
        const int t = s * (T_STEPS / SPLIT) + tt;

        // ---- stage-1 A fragment: rows r = t*NB + nb, K padded 24->32 ----
        const float* xr = x + ((long)t * NB + rowNb) * IN1;
        // prefetch next t-step's row of x (global_prefetch_b8)
        if (tt + 1 < T_STEPS / SPLIT) {
            __builtin_prefetch(xr + (long)NB * IN1, 0, 1);
        }
        const int koff = halfSel ? 16 : 0;
        const float4* p0 = (const float4*)(xr + koff);       // K koff..koff+7
        const float4  q0 = p0[0], q1 = p0[1];
        const float4* p1 = (const float4*)(xr + 8); // K 8..15 (used by h0) or dummy
        const float4  q2 = p1[0], q3 = p1[1];
        v16h a1;
        a1[0] = (f16)q0.x; a1[1] = (f16)q0.y; a1[2] = (f16)q0.z; a1[3] = (f16)q0.w;
        a1[4] = (f16)q1.x; a1[5] = (f16)q1.y; a1[6] = (f16)q1.z; a1[7] = (f16)q1.w;
        a1[8]  = (f16)(halfSel ? 0.0f : q2.x);
        a1[9]  = (f16)(halfSel ? 0.0f : q2.y);
        a1[10] = (f16)(halfSel ? 0.0f : q2.z);
        a1[11] = (f16)(halfSel ? 0.0f : q2.w);
        a1[12] = (f16)(halfSel ? 0.0f : q3.x);
        a1[13] = (f16)(halfSel ? 0.0f : q3.y);
        a1[14] = (f16)(halfSel ? 0.0f : q3.z);
        a1[15] = (f16)(halfSel ? 0.0f : q3.w);

        // ---- stage 1: 4 hTiles x (i,g,o) WMMA + activations ----
#pragma unroll
        for (int ht = 0; ht < 4; ++ht) {
            v16h bi = *(const v16h*)(w1f + ((0 * 4 + ht) * 32 + lane) * 16);
            v16h bg = *(const v16h*)(w1f + ((1 * 4 + ht) * 32 + lane) * 16);
            v16h bo = *(const v16h*)(w1f + ((2 * 4 + ht) * 32 + lane) * 16);
            v8f ci = __builtin_amdgcn_wmma_f32_16x16x32_f16(false, a1, false, bi, (short)0, zero, false, false);
            v8f cg = __builtin_amdgcn_wmma_f32_16x16x32_f16(false, a1, false, bg, (short)0, zero, false, false);
            v8f co = __builtin_amdgcn_wmma_f32_16x16x32_f16(false, a1, false, bo, (short)0, zero, false, false);
            const int ncol = ht * 16 + mrow;
            const float bi_s = b1s[ncol];
            const float bg_s = b1s[128 + ncol];
            const float bo_s = b1s[192 + ncol];
            const int mb = halfSel ? 8 : 0;
#pragma unroll
            for (int v = 0; v < 8; ++v) {
                const float iv = ci[v] + bi_s;
                const float gv = cg[v] + bg_s;
                const float ov = co[v] + bo_s;
                const float c  = fast_sigmoid(iv) * fast_tanh(gv);
                const float h  = fast_sigmoid(ov) * fast_tanh(c);
                stage[wave][mb + v][ncol] = (f16)h;
            }
        }
        asm volatile("s_wait_dscnt 0" ::: "memory");   // same-wave LDS relayout

        // ---- stage 2: ks = 2t, 2t+1 ; 6 N-tiles each ----
#pragma unroll
        for (int hi = 0; hi < 2; ++hi) {
            const int ks = 2 * t + hi;
            const int h0 = hi * 32 + (halfSel ? 16 : 0);
            v8h alo = *(const v8h*)&stage[wave][mrow][h0];        // K 16h+0..7
            v8h ahi = *(const v8h*)&stage[wave][mrow][h0 + 8];    // K 16h+8..15
            v16h a2 = __builtin_shufflevector(alo, ahi,
                        0, 1, 2, 3, 4, 5, 6, 7, 8, 9, 10, 11, 12, 13, 14, 15);
#pragma unroll
            for (int f = 0; f < 6; ++f) {
                v16h b2 = *(const v16h*)(w2f + (((long)ks * 6 + f) * 32 + lane) * 16);
                acc[f] = __builtin_amdgcn_wmma_f32_16x16x32_f16(false, a2, false, b2, (short)0, acc[f], false, false);
            }
        }
        asm volatile("s_wait_dscnt 0" ::: "memory");   // WAR vs next-iter stores
    }

    // ---- epilogue: reduce SPLIT partial accumulators ----
    {
        const int mb = halfSel ? 8 : 0;
#pragma unroll
        for (int f = 0; f < 6; ++f)
#pragma unroll
            for (int v = 0; v < 8; ++v)
                atomicAdd(&gates[grp][mb + v][f * 16 + mrow], acc[f][v]);
    }
    __syncthreads();

    // ---- note-cell activation + threshold, coalesced f32 store ----
    {
        const int laneInGroup = s * 32 + lane;   // 0..127
#pragma unroll
        for (int kk = 0; kk < 3; ++kk) {
            const int o = laneInGroup + kk * 128;       // 0..383
            const int m = o / 24, j = o % 24;
            const float iv = gates[grp][m][j];
            const float gv = gates[grp][m][48 + j];
            const float ov = gates[grp][m][72 + j];
            const float c  = fast_sigmoid(iv) * fast_tanh(gv);
            const float h  = fast_sigmoid(ov) * fast_tanh(c);
            out[(long)groupGlobal * 384 + o] = (h > 0.5f) ? 1.0f : 0.0f;
        }
    }
}

extern "C" void kernel_launch(void* const* d_in, const int* in_sizes, int n_in,
                              void* d_out, int out_size, void* d_ws, size_t ws_size,
                              hipStream_t stream)
{
    const float* x     = (const float*)d_in[0];
    const float* Wih_t = (const float*)d_in[1];
    // d_in[2] = W_hh_t (unused: zero state)
    const float* bih_t = (const float*)d_in[3];
    const float* bhh_t = (const float*)d_in[4];
    const float* Wih_n = (const float*)d_in[5];
    // d_in[6] = W_hh_n (unused: zero state)
    const float* bih_n = (const float*)d_in[7];
    const float* bhh_n = (const float*)d_in[8];

    char*  ws  = (char*)d_ws;
    f16*   w2f = (f16*)ws;
    f16*   w1f = (f16*)(ws + OFF_W1FRAG);
    float* b1s = (float*)(ws + OFF_B1);
    float* b2s = (float*)(ws + OFF_B2);
    float* out = (float*)d_out;

    prep_small<<<1, 256, 0, stream>>>(Wih_t, bih_t, bhh_t, bih_n, bhh_n, w1f, b1s, b2s);
    prep_w2<<<768, 256, 0, stream>>>(Wih_n, w2f);
    music_fused<<<NUM_BLOCKS, 256, 0, stream>>>(x, w2f, w1f, b1s, b2s, out);
}
